// Block_66812511256726
// MI455X (gfx1250) — matrix operand order repaired and checked by reading, not compiled, not run on hardware
//
#include <hip/hip_runtime.h>
#include <hip/hip_bf16.h>

// ---------------------------------------------------------------------------
// MI455X / gfx1250 transformer block.
// bf16 WMMA (v_wmma_f32_16x16x32_bf16, f32 accumulate) for all GEMMs and both
// attentions; flash-attention streams KV in 64-key chunks (8 WMMAs per online
// softmax update); self-attention mask realized as per-segment KV range
// (exact: exp(-10000+s) underflows to 0 in f32). Residual stream in f32 d_out.
// ---------------------------------------------------------------------------

typedef __bf16 bf16_t;
typedef __attribute__((ext_vector_type(16))) __bf16 v16bf;
typedef __attribute__((ext_vector_type(8)))  __bf16 v8bf;
typedef __attribute__((ext_vector_type(8)))  float  v8f;

#define NQ   2048
#define NKV  8192
#define CDIM 256
#define HEADS 8
#define LQSEG 256

__device__ __forceinline__ bf16_t f2bf(float f) {
  unsigned u = __builtin_bit_cast(unsigned, f);
  unsigned r = u + 0x7FFFu + ((u >> 16) & 1u);          // round-to-nearest-even
  unsigned short s = (unsigned short)(r >> 16);
  return __builtin_bit_cast(bf16_t, s);
}

__device__ __forceinline__ v16bf mk16(v8bf lo, v8bf hi) {
  v16bf r;
#pragma unroll
  for (int i = 0; i < 8; ++i) { r[i] = lo[i]; r[i + 8] = hi[i]; }
  return r;
}

// ------------------------------- f32 -> bf16 -------------------------------
__global__ void cvt_bf16(const float* __restrict__ in, bf16_t* __restrict__ out, int n) {
  int i = blockIdx.x * blockDim.x + threadIdx.x;
  if (i < n) out[i] = f2bf(in[i]);
}

// ---------------------- fused RMSNorm (+pos) -> bf16 -----------------------
__global__ void rmsnorm_kernel(const float* __restrict__ x, const float* __restrict__ w,
                               const float* __restrict__ pos,
                               bf16_t* __restrict__ out_pos, bf16_t* __restrict__ out_plain) {
  const int row = blockIdx.x, tid = threadIdx.x;
  const int idx = row * CDIM + tid;
  float v = x[idx];
  float ss = v * v;
#pragma unroll
  for (int msk = 1; msk < 32; msk <<= 1) ss += __shfl_xor(ss, msk, 32);
  __shared__ float sred[8];
  if ((tid & 31) == 0) sred[tid >> 5] = ss;
  __syncthreads();
  float tot = 0.f;
#pragma unroll
  for (int i = 0; i < 8; ++i) tot += sred[i];
  float rs = rsqrtf(tot * (1.0f / 256.0f) + 1.1920929e-7f);
  float nv = v * rs * w[tid];
  if (out_plain) out_plain[idx] = f2bf(nv);
  float pv = pos ? pos[idx] : 0.f;
  out_pos[idx] = f2bf(nv + pv);
}

// ------------------ transpose slice of bf16 matrix: out[c][r] ---------------
__global__ void transpose_cb(const bf16_t* __restrict__ in, int ldi, int col0,
                             bf16_t* __restrict__ out, int n) {
  int idx = blockIdx.x * blockDim.x + threadIdx.x;
  if (idx < n * CDIM) {
    int c = idx / n, r = idx % n;
    out[idx] = in[(size_t)r * ldi + col0 + c];
  }
}

// ------------------------------ WMMA GEMM ----------------------------------
// Y[M x N] = A[M x K](bf16) @ W[N x K]^T (bf16) (+bias)(+GELU)(+f32 residual)
// block = 256 threads = 8 waves (4 M x 2 N) -> 64(M) x 64(N) block tile.
// Each wave: 16x32 (two accumulators sharing one A fragment).
template<bool GELU, bool RES, bool OUTF, int K>
__global__ void gemm_wmma_kernel(const bf16_t* __restrict__ A, int lda,
                                 const bf16_t* __restrict__ W, int ldw,
                                 const float* __restrict__ bias,
                                 const float* __restrict__ res,
                                 float* __restrict__ outf, bf16_t* __restrict__ outb,
                                 int ldo) {
  const int lane = threadIdx.x & 31, wave = threadIdx.x >> 5;
  const int tm = blockIdx.x * 64 + (wave & 3) * 16;
  const int tn = blockIdx.y * 64 + (wave >> 2) * 32;
  const bf16_t* ap  = A + (size_t)(tm + (lane & 15)) * lda + ((lane >> 4) * 8);
  const bf16_t* wp0 = W + (size_t)(tn + (lane & 15)) * ldw + ((lane >> 4) * 8);
  const bf16_t* wp1 = wp0 + (size_t)16 * ldw;
  v8f acc0 = {0.f, 0.f, 0.f, 0.f, 0.f, 0.f, 0.f, 0.f};
  v8f acc1 = acc0;
#pragma unroll 4
  for (int k0 = 0; k0 < K; k0 += 32) {
    v16bf a  = mk16(*(const v8bf*)(ap + k0),  *(const v8bf*)(ap + k0 + 16));
    v16bf b0 = mk16(*(const v8bf*)(wp0 + k0), *(const v8bf*)(wp0 + k0 + 16));
    v16bf b1 = mk16(*(const v8bf*)(wp1 + k0), *(const v8bf*)(wp1 + k0 + 16));
    acc0 = __builtin_amdgcn_wmma_f32_16x16x32_bf16(false, a, false, b0, (short)0, acc0,
                                                   false, false);
    acc1 = __builtin_amdgcn_wmma_f32_16x16x32_bf16(false, a, false, b1, (short)0, acc1,
                                                   false, false);
  }
  const int n0 = tn + (lane & 15), n1 = n0 + 16;
  const float bv0 = bias ? bias[n0] : 0.f;
  const float bv1 = bias ? bias[n1] : 0.f;
#pragma unroll
  for (int r = 0; r < 8; ++r) {
    int m = tm + r + (lane >> 4) * 8;
    float v0 = acc0[r] + bv0;
    float v1 = acc1[r] + bv1;
    if (GELU) {
      v0 = 0.5f * v0 * (1.f + erff(v0 * 0.70710678118654752f));
      v1 = 0.5f * v1 * (1.f + erff(v1 * 0.70710678118654752f));
    }
    if (RES) {
      v0 += res[(size_t)m * ldo + n0];
      v1 += res[(size_t)m * ldo + n1];
    }
    if (OUTF) {
      outf[(size_t)m * ldo + n0] = v0;
      outf[(size_t)m * ldo + n1] = v1;
    } else {
      outb[(size_t)m * ldo + n0] = f2bf(v0);
      outb[(size_t)m * ldo + n1] = f2bf(v1);
    }
  }
}

// -------------------------- flash attention (WMMA) -------------------------
// One (head, 16-query tile) per wave; KV streamed in chunks of 64 keys:
// 4 score WMMAs + online softmax + LDS transpose of P + 4 PV WMMAs.
__global__ void attn_wmma_kernel(const bf16_t* __restrict__ Q, int ldq,
                                 const bf16_t* __restrict__ Km, int ldk,
                                 const bf16_t* __restrict__ Vt, int ldv,
                                 bf16_t* __restrict__ Out, int ldo,
                                 int nq, int mkv, int seg_q, int seg_kv, int masked) {
  __shared__ __align__(16) bf16_t plds[8 * 16 * 64];   // 16 KB
  const int lane = threadIdx.x & 31, wave = threadIdx.x >> 5;
  const int task = blockIdx.x * 8 + wave;
  const int nqt = nq >> 4;
  const int head = task / nqt;
  const int q0 = (task % nqt) << 4;
  const float scale = 0.17677669529663687f;  // D^-0.5, D=32

  // Q A-fragment (16 q-rows x 32 d), loaded once
  const bf16_t* qp = Q + (size_t)(q0 + (lane & 15)) * ldq + head * 32 + ((lane >> 4) * 8);
  const v16bf qa = mk16(*(const v8bf*)qp, *(const v8bf*)(qp + 16));

  int kv0 = 0, kv1 = mkv;
  if (masked) { int seg = q0 / seg_q; kv0 = seg * seg_kv; kv1 = kv0 + seg_kv; }

  const v8f z8 = {0.f, 0.f, 0.f, 0.f, 0.f, 0.f, 0.f, 0.f};
  v8f o0 = z8, o1 = z8;
  float mrow[8], lrow[8];
#pragma unroll
  for (int r = 0; r < 8; ++r) { mrow[r] = -1e30f; lrow[r] = 0.f; }

  bf16_t* lbase = plds + wave * 1024;
  const int mloc = (lane >> 4) * 8;
  const int col  = lane & 15;

  for (int kc = kv0; kc < kv1; kc += 64) {
    // ---- scores: four 16-key subtiles (d contiguous in memory) ----
    v8f s[4];
#pragma unroll
    for (int sub = 0; sub < 4; ++sub) {
      const bf16_t* kp = Km + (size_t)(kc + sub * 16 + (lane & 15)) * ldk +
                         head * 32 + ((lane >> 4) * 8);
      v16bf kb = mk16(*(const v8bf*)kp, *(const v8bf*)(kp + 16));
      s[sub] = __builtin_amdgcn_wmma_f32_16x16x32_bf16(false, qa, false, kb, (short)0, z8,
                                                       false, false);
    }
    // ---- online softmax (rows in VGPR index r, cols across 16 lanes) ----
#pragma unroll
    for (int r = 0; r < 8; ++r) {
      float a0 = s[0][r] * scale, a1 = s[1][r] * scale;
      float a2 = s[2][r] * scale, a3 = s[3][r] * scale;
      float t = fmaxf(fmaxf(a0, a1), fmaxf(a2, a3));
#pragma unroll
      for (int msk = 1; msk < 16; msk <<= 1) t = fmaxf(t, __shfl_xor(t, msk, 32));
      float mn = fmaxf(mrow[r], t);
      float c  = __expf(mrow[r] - mn);
      float e0 = __expf(a0 - mn), e1 = __expf(a1 - mn);
      float e2 = __expf(a2 - mn), e3 = __expf(a3 - mn);
      float rs = (e0 + e1) + (e2 + e3);
#pragma unroll
      for (int msk = 1; msk < 16; msk <<= 1) rs += __shfl_xor(rs, msk, 32);
      lrow[r] = lrow[r] * c + rs;
      mrow[r] = mn;
      o0[r] *= c; o1[r] *= c;
      // D-layout -> LDS (row-major P tile 16x64)
      bf16_t* lr = lbase + (mloc + r) * 64 + col;
      lr[0]  = f2bf(e0);
      lr[16] = f2bf(e1);
      lr[32] = f2bf(e2);
      lr[48] = f2bf(e3);
    }
    __syncthreads();
    // ---- reload P as two A-fragments (keys 0..31 and 32..63) ----
    const bf16_t* pp = lbase + (lane & 15) * 64 + ((lane >> 4) * 8);
    v16bf pa0 = mk16(*(const v8bf*)pp,        *(const v8bf*)(pp + 16));
    v16bf pa1 = mk16(*(const v8bf*)(pp + 32), *(const v8bf*)(pp + 48));
    // ---- V B-fragments (keys contiguous thanks to pre-transpose) ----
    const bf16_t* vrow0 = Vt + (size_t)(head * 32 + (lane & 15)) * ldv + kc + ((lane >> 4) * 8);
    const bf16_t* vrow1 = vrow0 + (size_t)16 * ldv;
    v16bf vb00 = mk16(*(const v8bf*)vrow0,        *(const v8bf*)(vrow0 + 16));
    v16bf vb01 = mk16(*(const v8bf*)(vrow0 + 32), *(const v8bf*)(vrow0 + 48));
    v16bf vb10 = mk16(*(const v8bf*)vrow1,        *(const v8bf*)(vrow1 + 16));
    v16bf vb11 = mk16(*(const v8bf*)(vrow1 + 32), *(const v8bf*)(vrow1 + 48));
    o0 = __builtin_amdgcn_wmma_f32_16x16x32_bf16(false, pa0, false, vb00, (short)0, o0,
                                                 false, false);
    o0 = __builtin_amdgcn_wmma_f32_16x16x32_bf16(false, pa1, false, vb01, (short)0, o0,
                                                 false, false);
    o1 = __builtin_amdgcn_wmma_f32_16x16x32_bf16(false, pa0, false, vb10, (short)0, o1,
                                                 false, false);
    o1 = __builtin_amdgcn_wmma_f32_16x16x32_bf16(false, pa1, false, vb11, (short)0, o1,
                                                 false, false);
    __syncthreads();
  }

#pragma unroll
  for (int r = 0; r < 8; ++r) {
    float inv = 1.f / lrow[r];
    int m = q0 + r + (lane >> 4) * 8;
    Out[(size_t)m * ldo + head * 32 + (lane & 15)]      = f2bf(o0[r] * inv);
    Out[(size_t)m * ldo + head * 32 + 16 + (lane & 15)] = f2bf(o1[r] * inv);
  }
}

// ---------------------------------------------------------------------------
extern "C" void kernel_launch(void* const* d_in, const int* in_sizes, int n_in,
                              void* d_out, int out_size, void* d_ws, size_t ws_size,
                              hipStream_t stream) {
  (void)in_sizes; (void)n_in; (void)out_size; (void)ws_size;
  const float* q        = (const float*)d_in[0];
  const float* kv       = (const float*)d_in[1];
  const float* pos_q    = (const float*)d_in[2];
  const float* pos_k    = (const float*)d_in[3];
  const float* w_nkv    = (const float*)d_in[4];
  const float* w_n1     = (const float*)d_in[5];
  const float* w_n2     = (const float*)d_in[6];
  const float* w_n3     = (const float*)d_in[7];
  const float* ca_wq    = (const float*)d_in[8];
  const float* ca_wk    = (const float*)d_in[9];
  const float* ca_wv    = (const float*)d_in[10];
  const float* ca_wo    = (const float*)d_in[11];
  const float* ca_bo    = (const float*)d_in[12];
  const float* sa_wq    = (const float*)d_in[13];
  const float* sa_wkv   = (const float*)d_in[14];
  const float* sa_wo    = (const float*)d_in[15];
  const float* sa_bo    = (const float*)d_in[16];
  const float* mlp_w1   = (const float*)d_in[17];
  const float* mlp_b1   = (const float*)d_in[18];
  const float* mlp_w2   = (const float*)d_in[19];
  const float* mlp_b2   = (const float*)d_in[20];
  float* xout = (float*)d_out;

  char* ws = (char*)d_ws;
  size_t off = 0;
  auto alloc = [&](size_t elems) -> bf16_t* {
    bf16_t* p = (bf16_t*)(ws + off); off += elems * sizeof(bf16_t); return p;
  };
  // bf16 weights
  bf16_t* wqb   = alloc(65536);
  bf16_t* wkb   = alloc(65536);
  bf16_t* wvb   = alloc(65536);
  bf16_t* wob   = alloc(65536);
  bf16_t* swqb  = alloc(65536);
  bf16_t* swkvb = alloc(131072);
  bf16_t* swob  = alloc(65536);
  bf16_t* w1b   = alloc(262144);
  bf16_t* w2b   = alloc(262144);
  // bf16 activations (lifetime-aliased; ~26 MB total)
  bf16_t* kvnp = alloc(2097152);   // rmsnorm(kv)+pos_k
  bf16_t* kvn  = alloc(2097152);   // rmsnorm(kv)
  bf16_t* qnp  = alloc(524288);    // rmsnorm(q)+pos_q
  bf16_t* qc   = alloc(524288);
  bf16_t* kc   = alloc(2097152);
  bf16_t* vc   = alloc(2097152);
  bf16_t* vct  = alloc(2097152);   // V^T for cross attn
  bf16_t* feat = alloc(524288);
  // aliases into dead regions
  bf16_t* xn2p  = kvnp;            // dead after kc GEMM
  bf16_t* xn2   = kvnp + 524288;
  bf16_t* qs    = kvnp + 1048576;
  bf16_t* kvs   = kvn;             // dead after vc GEMM
  bf16_t* vst   = kvn + 1048576;
  bf16_t* feat2 = qnp;             // dead after qc GEMM
  bf16_t* xn3   = qc;              // dead after cross attn
  bf16_t* hb    = vc;              // dead after V transpose

  auto cvt = [&](const float* s, bf16_t* d, int n) {
    cvt_bf16<<<dim3((n + 255) / 256), dim3(256), 0, stream>>>(s, d, n);
  };
  cvt(ca_wq, wqb, 65536);   cvt(ca_wk, wkb, 65536);   cvt(ca_wv, wvb, 65536);
  cvt(ca_wo, wob, 65536);   cvt(sa_wq, swqb, 65536);  cvt(sa_wkv, swkvb, 131072);
  cvt(sa_wo, swob, 65536);  cvt(mlp_w1, w1b, 262144); cvt(mlp_w2, w2b, 262144);

  // ---- cross-attention branch ----
  rmsnorm_kernel<<<dim3(NKV), dim3(256), 0, stream>>>(kv, w_nkv, pos_k, kvnp, kvn);
  rmsnorm_kernel<<<dim3(NQ),  dim3(256), 0, stream>>>(q,  w_n1,  pos_q, qnp, nullptr);

  gemm_wmma_kernel<false,false,false,256><<<dim3(NQ/64, 4), dim3(256), 0, stream>>>(
      qnp, 256, wqb, 256, nullptr, nullptr, nullptr, qc, 256);
  gemm_wmma_kernel<false,false,false,256><<<dim3(NKV/64, 4), dim3(256), 0, stream>>>(
      kvnp, 256, wkb, 256, nullptr, nullptr, nullptr, kc, 256);
  gemm_wmma_kernel<false,false,false,256><<<dim3(NKV/64, 4), dim3(256), 0, stream>>>(
      kvn, 256, wvb, 256, nullptr, nullptr, nullptr, vc, 256);
  transpose_cb<<<dim3((NKV * CDIM + 255) / 256), dim3(256), 0, stream>>>(vc, 256, 0, vct, NKV);

  attn_wmma_kernel<<<dim3(HEADS * (NQ / 16) / 8), dim3(256), 0, stream>>>(
      qc, 256, kc, 256, vct, NKV, feat, 256, NQ, NKV, LQSEG, LQSEG, 0);

  gemm_wmma_kernel<false,true,true,256><<<dim3(NQ/64, 4), dim3(256), 0, stream>>>(
      feat, 256, wob, 256, ca_bo, q, xout, nullptr, 256);

  // ---- self-attention branch ----
  rmsnorm_kernel<<<dim3(NQ), dim3(256), 0, stream>>>(xout, w_n2, pos_q, xn2p, xn2);
  gemm_wmma_kernel<false,false,false,256><<<dim3(NQ/64, 4), dim3(256), 0, stream>>>(
      xn2p, 256, swqb, 256, nullptr, nullptr, nullptr, qs, 256);
  gemm_wmma_kernel<false,false,false,256><<<dim3(NQ/64, 8), dim3(256), 0, stream>>>(
      xn2, 256, swkvb, 256, nullptr, nullptr, nullptr, kvs, 512);
  transpose_cb<<<dim3((NQ * CDIM + 255) / 256), dim3(256), 0, stream>>>(kvs, 512, 256, vst, NQ);

  attn_wmma_kernel<<<dim3(HEADS * (NQ / 16) / 8), dim3(256), 0, stream>>>(
      qs, 256, kvs, 512, vst, NQ, feat2, 256, NQ, NQ, LQSEG, LQSEG, 1);

  gemm_wmma_kernel<false,true,true,256><<<dim3(NQ/64, 4), dim3(256), 0, stream>>>(
      feat2, 256, swob, 256, sa_bo, xout, xout, nullptr, 256);

  // ---- MLP ----
  rmsnorm_kernel<<<dim3(NQ), dim3(256), 0, stream>>>(xout, w_n3, nullptr, xn3, nullptr);
  gemm_wmma_kernel<true,false,false,256><<<dim3(NQ/64, 16), dim3(256), 0, stream>>>(
      xn3, 256, w1b, 256, mlp_b1, nullptr, nullptr, hb, 1024);
  gemm_wmma_kernel<false,true,true,1024><<<dim3(NQ/64, 4), dim3(256), 0, stream>>>(
      hb, 1024, w2b, 1024, mlp_b2, xout, xout, nullptr, 256);
}
